// ResiDual_90400471646858
// MI455X (gfx1250) — matrix operand order repaired and checked by reading, not compiled, not run on hardware
//
#include <hip/hip_runtime.h>
#include <hip/hip_bf16.h>
#include <stdint.h>

// Problem constants (from reference)
#define Nn 1024
#define Ss 144
#define Rr 96
#define Kk 128
#define Dd 768
#define NABL (Ss - Rr)   // 48

typedef __bf16 v16bf __attribute__((ext_vector_type(16)));
typedef float  v8f   __attribute__((ext_vector_type(8)));

// ---------------------------------------------------------------------------
// CDNA5 async global->LDS copy (ASYNCcnt tracked), 16 bytes per lane.
// lds_byte_addr: low 32 bits of the generic LDS address (== LDS offset).
// ---------------------------------------------------------------------------
__device__ inline void async_copy_b128(uint32_t lds_byte_addr,
                                       const void* gptr) {
  asm volatile("global_load_async_to_lds_b128 %0, %1, off"
               :
               : "v"(lds_byte_addr), "v"((uint64_t)(uintptr_t)gptr)
               : "memory");
}

__device__ inline void wait_async0() {
  asm volatile("s_wait_asynccnt 0" ::: "memory");
}

// ---------------------------------------------------------------------------
// Fragment loaders per CDNA5 ISA VGPR layouts (CDNA5 ISA §7.12.2)
// A operand (16x32 bf16): lane m = lane&15 holds row M=m.
//   VGPR v holds k-pair at ((v<4)? 2v : 16+2(v-4)) + 8*laneHi.
// B operand (32x16 bf16): lane holds column N = lane&15.
//   VGPR v holds k-pair at ((v<4)? 2v : 16+2(v-4)) + 8*laneHi (same as A).
// All pair offsets are even -> 4B aligned; compiler merges into ds_load_b128.
// ---------------------------------------------------------------------------
__device__ inline v16bf load_frag_A(const __bf16* lds, int row, int stride,
                                    int kbase, int hi) {
  v16bf f;
  const __bf16* p = lds + row * stride + kbase + 8 * hi;
#pragma unroll
  for (int v = 0; v < 8; ++v) {
    int k0 = (v < 4) ? (2 * v) : (16 + 2 * (v - 4));
    union { uint32_t u; __bf16 h[2]; } cv;
    cv.u = *(const uint32_t*)(p + k0);
    f[2 * v]     = cv.h[0];
    f[2 * v + 1] = cv.h[1];
  }
  return f;
}

__device__ inline v16bf load_frag_B(const __bf16* lds, int row, int stride,
                                    int kbase, int hi) {
  v16bf f;
  const __bf16* p = lds + row * stride + kbase + 8 * hi;
#pragma unroll
  for (int v = 0; v < 8; ++v) {
    int k0 = (v < 4) ? (2 * v) : (16 + 2 * (v - 4));
    union { uint32_t u; __bf16 h[2]; } cv;
    cv.u = *(const uint32_t*)(p + k0);
    f[2 * v]     = cv.h[0];
    f[2 * v + 1] = cv.h[1];
  }
  return f;
}

// ---------------------------------------------------------------------------
// Prep kernel (per launch): bf16-convert the r-reused operands into ws.
//   bs_ws[r][k][d] = bf16(lambda[r,k] * bases[r,k,d])
//   pt_ws[r][d][k] = bf16(inv_bases[r,k,d])      (transposed via LDS tile)
// grid = (R, D/128), block = 256.
// ---------------------------------------------------------------------------
__global__ __launch_bounds__(256) void prep_bases_kernel(
    const float* __restrict__ bases, const float* __restrict__ inv_bases,
    const float* __restrict__ lambdas, const int* __restrict__ lengths,
    __bf16* __restrict__ bs_ws, __bf16* __restrict__ pt_ws) {
  __shared__ float  lam[Kk];
  __shared__ __bf16 tile[128 * 129];   // padded transpose staging

  const int tid = threadIdx.x;
  const int r   = blockIdx.x;
  const int d0  = blockIdx.y * 128;

  // ragged lambda -> padded row (uniform scalar cumsum over lengths)
  int off = 0;
  for (int i = 0; i < r; ++i) off += lengths[i];
  if (tid < Kk) {
    int len = lengths[r];
    lam[tid] = (tid < len) ? lambdas[off + tid] : 0.0f;
  }
  __syncthreads();

#pragma unroll 4
  for (int i = 0; i < 64; ++i) {
    int e = i * 256 + tid;
    int k = e >> 7, d = e & 127;
    size_t gi = ((size_t)r * Kk + k) * Dd + d0 + d;
    bs_ws[gi] = (__bf16)(bases[gi] * lam[k]);
    tile[k * 129 + d] = (__bf16)inv_bases[gi];
  }
  __syncthreads();
#pragma unroll 4
  for (int i = 0; i < 64; ++i) {
    int e  = i * 256 + tid;
    int dr = e >> 7, k = e & 127;
    pt_ws[((size_t)r * Dd + d0 + dr) * Kk + k] = tile[k * 129 + dr];
  }
}

// ---------------------------------------------------------------------------
// Kernel 0: out[n,d] = sum over ablated slices of x (initializes out before
// the atomic accumulation of the residual reconstruction).
// ---------------------------------------------------------------------------
__global__ __launch_bounds__(256) void ablate_init_kernel(
    const float* __restrict__ x, const int* __restrict__ abl,
    float* __restrict__ out) {
  int idx = blockIdx.x * blockDim.x + threadIdx.x;
  if (idx >= Nn * Dd) return;
  int n = idx / Dd;
  int d = idx - n * Dd;
  float s = 0.0f;
#pragma unroll 4
  for (int j = 0; j < NABL; ++j) {
    s += x[(size_t)n * (Ss * Dd) + (size_t)abl[j] * Dd + d];
  }
  out[idx] = s;
}

// ---------------------------------------------------------------------------
// Main kernel: one block = (64-row N-tile, one residual stream r).
// Phase 1: coeffs(64x128) = (x_tile - mean_r) @ (lambda * B_r)^T  [bf16 WMMA]
// Phase 2: out += coeffs @ P_r                                    [bf16 WMMA]
// B-operand tiles staged with global_load_async_to_lds_b128 from bf16 ws.
// ---------------------------------------------------------------------------
__global__ __launch_bounds__(256) void residual_wmma_kernel(
    const float* __restrict__ x, const float* __restrict__ means,
    const int* __restrict__ res_idx, const __bf16* __restrict__ bs_ws,
    const __bf16* __restrict__ pt_ws, float* __restrict__ out) {
  // 64x128 bf16 (phase1: adapted x chunk; phase2: coeffs) = 16 KB
  __shared__ __align__(16) __bf16 ldsA[64 * 128];
  // 128x128 bf16 (phase1: scaled bases [k][d]; phase2: invbases^T [d][k])
  __shared__ __align__(16) __bf16 ldsB[128 * 128];   // 32 KB

  const int tid  = threadIdx.x;
  const int lane = tid & 31;
  const int w    = tid >> 5;       // wave id 0..7
  const int hi   = lane >> 4;      // lane half
  const int ln   = lane & 15;
  const int mbase  = (w & 3) * 16;   // wave's M tile (rows within 64)
  const int ktbase = (w >> 2) * 4;   // wave's 4 N-tiles (cols, 16 each)

  const int r  = blockIdx.y;
  const int n0 = blockIdx.x * 64;
  const int sres = res_idx[r];

  const uint32_t ldsB_addr = (uint32_t)(uintptr_t)(&ldsB[0]);

  // ---------------- Phase 1: coeffs = A @ Bs^T --------------------------
  v8f acc[4];
#pragma unroll
  for (int j = 0; j < 4; ++j)
#pragma unroll
    for (int jj = 0; jj < 8; ++jj) acc[j][jj] = 0.0f;

  for (int dc = 0; dc < Dd / 128; ++dc) {
    const int d0 = dc * 128;
    __syncthreads();   // previous iteration's LDS reads complete
    // async-stage lambda-scaled bases tile -> ldsB[k][0..127]
    {
      const __bf16* src = bs_ws + (size_t)r * Kk * Dd + d0;
      const int col = (tid & 15) * 8;          // bf16 elems (16B chunk)
#pragma unroll
      for (int i = 0; i < 8; ++i) {
        int row = i * 16 + (tid >> 4);
        async_copy_b128(ldsB_addr + (uint32_t)(row * 256 + (tid & 15) * 16),
                        src + (size_t)row * Dd + col);
      }
    }
    // stage adapted x tile -> ldsA[n][d] (bf16, VALU convert path)
#pragma unroll 4
    for (int i = 0; i < 32; ++i) {
      int e = i * 256 + tid;
      int n = e >> 7, d = e & 127;
      float v = x[(size_t)(n0 + n) * (Ss * Dd) + (size_t)sres * Dd + d0 + d] -
                means[r * Dd + d0 + d];
      ldsA[n * 128 + d] = (__bf16)v;
    }
    wait_async0();     // this wave's async LDS writes done
    __syncthreads();   // whole tile staged across waves
#pragma unroll
    for (int kk = 0; kk < 4; ++kk) {        // 32-wide K (=d) steps
      const int dl = kk * 32;
      v16bf a = load_frag_A(ldsA, mbase + ln, 128, dl, hi);
#pragma unroll
      for (int j = 0; j < 4; ++j) {
        v16bf b = load_frag_B(ldsB, (ktbase + j) * 16 + ln, 128, dl, hi);
        acc[j] = __builtin_amdgcn_wmma_f32_16x16x32_bf16(
            false, a, false, b, (short)0, acc[j], false, false);
      }
    }
  }
  __syncthreads();
  // coeffs (f32 accum) -> bf16 into ldsA[n][k]
#pragma unroll
  for (int j = 0; j < 4; ++j) {
    int kcol = (ktbase + j) * 16 + ln;
#pragma unroll
    for (int jj = 0; jj < 8; ++jj) {
      int n = mbase + jj + 8 * hi;          // C/D layout: M = vgpr + 8*laneHi
      ldsA[n * 128 + kcol] = (__bf16)acc[j][jj];
    }
  }
  __syncthreads();

  // ---------------- Phase 2: out += coeffs @ P --------------------------
  for (int dc = 0; dc < Dd / 128; ++dc) {
    if (dc) __syncthreads();
    // async-stage inv_bases^T tile (contiguous 32 KB) -> ldsB[d][k]
    {
      const __bf16* src = pt_ws + ((size_t)r * Dd + dc * 128) * Kk;
#pragma unroll
      for (int i = 0; i < 8; ++i) {
        int e = i * 256 + tid;               // 16B chunk id
        async_copy_b128(ldsB_addr + (uint32_t)e * 16, src + (size_t)e * 8);
      }
    }
    wait_async0();
    __syncthreads();
    v8f acc2[4];
#pragma unroll
    for (int j = 0; j < 4; ++j)
#pragma unroll
      for (int jj = 0; jj < 8; ++jj) acc2[j][jj] = 0.0f;
#pragma unroll
    for (int kk = 0; kk < 4; ++kk) {        // 32-wide K steps over Kk=128
      const int kl = kk * 32;
      v16bf a = load_frag_A(ldsA, mbase + ln, 128, kl, hi);
#pragma unroll
      for (int j = 0; j < 4; ++j) {
        v16bf b = load_frag_B(ldsB, (ktbase + j) * 16 + ln, 128, kl, hi);
        acc2[j] = __builtin_amdgcn_wmma_f32_16x16x32_bf16(
            false, a, false, b, (short)0, acc2[j], false, false);
      }
    }
    // accumulate over r via f32 atomics into out (L2-resident, 3 MB)
#pragma unroll
    for (int j = 0; j < 4; ++j) {
      int d = dc * 128 + (ktbase + j) * 16 + ln;
#pragma unroll
      for (int jj = 0; jj < 8; ++jj) {
        int n = n0 + mbase + jj + 8 * hi;
        atomicAdd(&out[(size_t)n * Dd + d], acc2[j][jj]);
      }
    }
  }
}

extern "C" void kernel_launch(void* const* d_in, const int* in_sizes, int n_in,
                              void* d_out, int out_size, void* d_ws,
                              size_t ws_size, hipStream_t stream) {
  (void)in_sizes; (void)n_in; (void)out_size; (void)ws_size;
  const float* x         = (const float*)d_in[0];
  const float* lambdas   = (const float*)d_in[1];
  const float* bases     = (const float*)d_in[2];
  const float* inv_bases = (const float*)d_in[3];
  const float* means     = (const float*)d_in[4];
  const int*   res_idx   = (const int*)d_in[5];
  const int*   abl_idx   = (const int*)d_in[6];
  const int*   lengths   = (const int*)d_in[7];
  float*       out       = (float*)d_out;

  // ws layout: [ bs_ws : R*K*D bf16 | pt_ws : R*D*K bf16 ]  (~37.7 MB)
  __bf16* bs_ws = (__bf16*)d_ws;
  __bf16* pt_ws = bs_ws + (size_t)Rr * Kk * Dd;

  // Pass 0: bf16-convert lambda-scaled bases + transposed inv_bases into ws.
  prep_bases_kernel<<<dim3(Rr, Dd / 128), 256, 0, stream>>>(
      bases, inv_bases, lambdas, lengths, bs_ws, pt_ws);
  // Pass 1: initialize out with the ablated-slice sum.
  ablate_init_kernel<<<(Nn * Dd + 255) / 256, 256, 0, stream>>>(x, abl_idx,
                                                                out);
  // Pass 2: WMMA residual reconstruction, atomically accumulated over r.
  dim3 grid(Nn / 64, Rr);
  residual_wmma_kernel<<<grid, 256, 0, stream>>>(x, means, res_idx, bs_ws,
                                                 pt_ws, out);
}
